// Transformer_120259084773
// MI455X (gfx1250) — matrix operand order repaired and checked by reading, not compiled, not run on hardware
//
#include <hip/hip_runtime.h>
#include <hip/hip_bf16.h>

// ---------------------------------------------------------------------------
// Transformer encoder-decoder forward for MI455X (gfx1250, wave32, WMMA).
// GEMMs: f16 operands (A [M,K], BT [N,K]) with f32 accumulate via
// v_wmma_f32_16x16x32_f16. Tiles exactly divide all shapes -> no bounds
// checks, EXEC all-1s. Staging uses GLOBAL_LOAD_ASYNC_TO_LDS_B128
// (ASYNCcnt path; builtin confirmed present, signature: (global v4i*,
// local v4i*, imm offset, imm cpol)), else 128-bit sync copies.
// Softmax / LayerNorm / masking / PE stay fp32.
// ---------------------------------------------------------------------------

typedef _Float16 v16h __attribute__((ext_vector_type(16)));
typedef _Float16 h8   __attribute__((ext_vector_type(8)));
typedef float    v8f  __attribute__((ext_vector_type(8)));
typedef int      v4i  __attribute__((ext_vector_type(4)));

#define BM 128
#define BN 64
#define BKK 32
#define LDT 40  // LDS row stride in halves (80B: 16B-aligned, skews banks)

#if __has_builtin(__builtin_amdgcn_global_load_async_to_lds_b128)
#define ASYNC_LDS 1
#endif

typedef __attribute__((address_space(1))) v4i* g4p;  // global int4*
typedef __attribute__((address_space(3))) v4i* l4p;  // LDS int4*

__device__ __forceinline__ void stage16(const _Float16* g, _Float16* l) {
#ifdef ASYNC_LDS
  // generic->AS1: generic address of global memory == AS1 address.
  // generic->AS3: low 32 bits of a generic LDS address == LDS offset.
  g4p gp = (g4p)(unsigned long long)g;
  l4p lp = (l4p)(unsigned)(unsigned long long)l;
  __builtin_amdgcn_global_load_async_to_lds_b128(gp, lp, 0, 0);
#else
  *(h8*)l = *(const h8*)g;
#endif
}
__device__ __forceinline__ void stage_wait() {
#ifdef ASYNC_LDS
#if __has_builtin(__builtin_amdgcn_s_wait_asynccnt)
  __builtin_amdgcn_s_wait_asynccnt(0);
#else
  asm volatile("s_wait_asynccnt 0x0" ::: "memory");
#endif
#endif
}

// -------------------------- wave32 reductions ------------------------------
__device__ __forceinline__ float wred_max(float v) {
#pragma unroll
  for (int o = 16; o > 0; o >>= 1) v = fmaxf(v, __shfl_xor(v, o, 32));
  return v;
}
__device__ __forceinline__ float wred_sum(float v) {
#pragma unroll
  for (int o = 16; o > 0; o >>= 1) v += __shfl_xor(v, o, 32);
  return v;
}

// ------------------------------ WMMA GEMM ----------------------------------
// C[z] = A[z] (MxK f16) * BT[z]^T (BT NxK f16) [+bias] [relu]; opt f16 mirror.
// Requirements: M%128==0, N%64==0, K%32==0.
// Block: 256 threads = 8 waves (4x2), each wave a 32x32 patch (2x2 WMMA).
__global__ __launch_bounds__(256) void wmma_gemm_kernel(
    const _Float16* __restrict__ A, const _Float16* __restrict__ BT,
    const float* __restrict__ bias, float* __restrict__ C,
    _Float16* __restrict__ C16, int M, int N, int K,
    long long sA, long long sB, long long sC, int relu) {
  __shared__ _Float16 As[BM * LDT];
  __shared__ _Float16 Bs[BN * LDT];

  const long long z = blockIdx.z;
  A += z * sA;
  BT += z * sB;
  const long long cbase = z * sC;

  const int tid = threadIdx.x;
  const int lane = tid & 31;
  const int wave = tid >> 5;
  const int wm = (wave & 3) * 32;
  const int wn = (wave >> 2) * 32;
  const int m0 = blockIdx.y * BM;
  const int n0 = blockIdx.x * BN;

  v8f acc[2][2] = {};

  const int lr = lane & 15;
  const int hi = lane >> 4;  // 0 | 1
  const int kbA = hi << 3;   // A frag k-base: 0 | 8
  const int kbB = hi << 3;   // B frag k-base: 0 | 8

  // staging: 16B chunks; A tile 128x32 halves -> 2 chunks/thread,
  // B tile 64x32 halves -> 1 chunk/thread.
  const int srow = tid >> 2;        // 0..63
  const int scol = (tid & 3) << 3;  // 0,8,16,24 (halves)

  const _Float16* Ag = A + (long long)m0 * K + (long long)srow * K + scol;
  const _Float16* Bg = BT + (long long)n0 * K + (long long)srow * K + scol;
  const long long a64 = (long long)64 * K;
  _Float16* la0 = &As[srow * LDT + scol];
  _Float16* la1 = &As[(srow + 64) * LDT + scol];
  _Float16* lb = &Bs[srow * LDT + scol];

  for (int k0 = 0; k0 < K; k0 += BKK) {
    stage16(Ag + k0, la0);
    stage16(Ag + a64 + k0, la1);
    stage16(Bg + k0, lb);
    stage_wait();
    __syncthreads();

    v16h af[2], bf[2];
#pragma unroll
    for (int t = 0; t < 2; ++t) {
      const _Float16* ap = &As[(wm + t * 16 + lr) * LDT];
      h8 a0 = *(const h8*)(ap + kbA);       // K {0..7} | {8..15}
      h8 a1 = *(const h8*)(ap + 16 + kbA);  // K {16..23} | {24..31}
      af[t] = __builtin_shufflevector(a0, a1, 0, 1, 2, 3, 4, 5, 6, 7, 8, 9, 10,
                                      11, 12, 13, 14, 15);
      const _Float16* bp = &Bs[(wn + t * 16 + lr) * LDT];
      h8 b0 = *(const h8*)(bp + kbB);      // K {0..7} | {8..15}
      h8 b1 = *(const h8*)(bp + 16 + kbB);  // K {16..23} | {24..31}
      bf[t] = __builtin_shufflevector(b0, b1, 0, 1, 2, 3, 4, 5, 6, 7, 8, 9, 10,
                                      11, 12, 13, 14, 15);
    }

#pragma unroll
    for (int tm = 0; tm < 2; ++tm)
#pragma unroll
      for (int tn = 0; tn < 2; ++tn)
        acc[tm][tn] = __builtin_amdgcn_wmma_f32_16x16x32_f16(
            false, af[tm], false, bf[tn], (short)0, acc[tm][tn], false, false);
    __syncthreads();
  }

  // epilogue: C/D layout row = vgpr + 8*(lane>=16), col = lane&15
  const int rofs = hi << 3;
  float bv[2] = {0.f, 0.f};
  if (bias) {
    bv[0] = bias[n0 + wn + lr];
    bv[1] = bias[n0 + wn + 16 + lr];
  }
#pragma unroll
  for (int tm = 0; tm < 2; ++tm)
#pragma unroll
    for (int tn = 0; tn < 2; ++tn)
#pragma unroll
      for (int r = 0; r < 8; ++r) {
        float v = acc[tm][tn][r] + bv[tn];
        if (relu) v = fmaxf(v, 0.f);
        acc[tm][tn][r] = v;
      }
#pragma unroll
  for (int tm = 0; tm < 2; ++tm)
#pragma unroll
    for (int tn = 0; tn < 2; ++tn)
#pragma unroll
      for (int r = 0; r < 8; ++r) {
        long long row = m0 + wm + tm * 16 + r + rofs;
        long long col = n0 + wn + tn * 16 + lr;
        C[cbase + row * N + col] = acc[tm][tn][r];
      }
  if (C16) {
#pragma unroll
    for (int tm = 0; tm < 2; ++tm)
#pragma unroll
      for (int tn = 0; tn < 2; ++tn)
#pragma unroll
        for (int r = 0; r < 8; ++r) {
          long long row = m0 + wm + tm * 16 + r + rofs;
          long long col = n0 + wn + tn * 16 + lr;
          C16[cbase + row * N + col] = (_Float16)acc[tm][tn][r];
        }
  }
}

// ---------------- f32 [K,N] -> f16 [N,K] transpose-convert -----------------
// K = 1<<kshift (512 or 2048 here), so div/mod are shifts.
__global__ void cvtT_kernel(const float* __restrict__ in,
                            _Float16* __restrict__ out, int kshift, int N,
                            long long total) {
  long long i = (long long)blockIdx.x * blockDim.x + threadIdx.x;
  if (i >= total) return;
  int k = (int)(i & ((1 << kshift) - 1));
  long long n = i >> kshift;
  out[i] = (_Float16)in[(long long)k * N + n];
}

// ------------------------ masked softmax -----------------------------------
// sc (f32 scratch, in-place) : [B*H, 512, 512]; probs -> o16 (f16).
// One block (256 thr) per row. PAD id == 1. Fully-masked rows -> 0.
__global__ __launch_bounds__(256) void softmax_mask_kernel(
    float* __restrict__ sc, _Float16* __restrict__ o16,
    const int* __restrict__ tok, int causal, float scale) {
  const int SK = 512;
  const long long row = blockIdx.x;
  const int q = (int)(row & 511);
  const int b = (int)(row >> 12);  // H*Sq = 8*512 = 4096
  float* x = sc + row * SK;
  _Float16* y = o16 + row * SK;
  const int* tr = tok + (long long)b * SK;
  const int tid = threadIdx.x, lane = tid & 31, wv = tid >> 5;
  __shared__ float sred[8];
  const float NEG_INF = -__builtin_inff();

  float mx = NEG_INF;
#pragma unroll
  for (int c = tid; c < SK; c += 256) {
    bool m = (tr[c] == 1) || (causal && c > q);
    if (!m) mx = fmaxf(mx, x[c] * scale);
  }
  mx = wred_max(mx);
  if (!lane) sred[wv] = mx;
  __syncthreads();
  if (tid < 32) {
    float t = (lane < 8) ? sred[lane] : NEG_INF;
    t = wred_max(t);
    if (!lane) sred[0] = t;
  }
  __syncthreads();
  mx = sred[0];
  __syncthreads();

  float sum = 0.f;
#pragma unroll
  for (int c = tid; c < SK; c += 256) {
    bool m = (tr[c] == 1) || (causal && c > q);
    float e = (m || mx == NEG_INF) ? 0.f : expf(x[c] * scale - mx);
    x[c] = e;
    sum += e;
  }
  sum = wred_sum(sum);
  if (!lane) sred[wv] = sum;
  __syncthreads();
  if (tid < 32) {
    float t = (lane < 8) ? sred[lane] : 0.f;
    t = wred_sum(t);
    if (!lane) sred[0] = t;
  }
  __syncthreads();
  const float inv = (sred[0] > 0.f) ? 1.f / sred[0] : 0.f;  // nan_to_num
#pragma unroll
  for (int c = tid; c < SK; c += 256) y[c] = (_Float16)(x[c] * inv);
}

// --------------------- residual add + LayerNorm (D=512) --------------------
__global__ __launch_bounds__(256) void add_ln_kernel(
    const float* __restrict__ x, const float* __restrict__ a,
    const float* __restrict__ g, const float* __restrict__ bb,
    float* __restrict__ out, _Float16* __restrict__ out16) {
  const int D = 512;
  const long long row = blockIdx.x;
  const float* xr = x + row * D;
  const int tid = threadIdx.x, lane = tid & 31, wv = tid >> 5;
  __shared__ float sred[8];

  float v0 = xr[tid], v1 = xr[tid + 256];
  if (a) {
    const float* ar = a + row * D;
    v0 += ar[tid];
    v1 += ar[tid + 256];
  }
  float s = wred_sum(v0 + v1);
  if (!lane) sred[wv] = s;
  __syncthreads();
  if (tid < 32) {
    float t = (lane < 8) ? sred[lane] : 0.f;
    t = wred_sum(t);
    if (!lane) sred[0] = t;
  }
  __syncthreads();
  const float mean = sred[0] / D;
  __syncthreads();

  const float d0 = v0 - mean, d1 = v1 - mean;
  s = wred_sum(d0 * d0 + d1 * d1);
  if (!lane) sred[wv] = s;
  __syncthreads();
  if (tid < 32) {
    float t = (lane < 8) ? sred[lane] : 0.f;
    t = wred_sum(t);
    if (!lane) sred[0] = t;
  }
  __syncthreads();
  const float inv = rsqrtf(sred[0] / D + 1e-5f);

  const float r0 = g[tid] * d0 * inv + bb[tid];
  const float r1 = g[tid + 256] * d1 * inv + bb[tid + 256];
  float* orow = out + row * D;
  orow[tid] = r0;
  orow[tid + 256] = r1;
  if (out16) {
    _Float16* o16 = out16 + row * D;
    o16[tid] = (_Float16)r0;
    o16[tid + 256] = (_Float16)r1;
  }
}

// ---------------- embedding lookup * sqrt(D) + sinusoidal PE ---------------
// D = 512, S = 512 hardcoded (shifts instead of div/mod).
__global__ void embed_kernel(const float* __restrict__ emb,
                             const int* __restrict__ tok,
                             float* __restrict__ out,
                             _Float16* __restrict__ out16, long long total) {
  long long i = (long long)blockIdx.x * blockDim.x + threadIdx.x;
  if (i >= total) return;
  int d = (int)(i & 511);
  long long bs = i >> 9;
  int s = (int)(bs & 511);
  int t = tok[bs];
  int i2 = d & ~1;
  float div = expf(-0.0179889463f * (float)i2);  // -ln(10000)/512 * i2
  float ang = (float)s * div;
  float pe = (d & 1) ? cosf(ang) : sinf(ang);
  float v = emb[((long long)t << 9) + d] * 22.6274169980f + pe;  // sqrt(512)
  out[i] = v;
  out16[i] = (_Float16)v;
}

// --------------------------- head permutations -----------------------------
// Fixed dims: B=4, H=8, S=512, dk=64 -> all shifts/masks.
// [B,S,512] f32 -> [B,H,S,dk] f16
__global__ void split_heads_f16_kernel(const float* __restrict__ in,
                                       _Float16* __restrict__ out,
                                       long long tot) {
  long long i = (long long)blockIdx.x * blockDim.x + threadIdx.x;
  if (i >= tot) return;
  int j = (int)(i & 63);
  int s = (int)((i >> 6) & 511);
  int h = (int)((i >> 15) & 7);
  long long b = i >> 18;
  out[i] = (_Float16)in[(((b << 9) + s) << 9) + (h << 6) + j];
}
// [B,S,512] f32 -> [B,H,dk,S] f16
__global__ void split_headsT_f16_kernel(const float* __restrict__ in,
                                        _Float16* __restrict__ out,
                                        long long tot) {
  long long i = (long long)blockIdx.x * blockDim.x + threadIdx.x;
  if (i >= tot) return;
  int s = (int)(i & 511);
  int j = (int)((i >> 9) & 63);
  int h = (int)((i >> 15) & 7);
  long long b = i >> 18;
  out[i] = (_Float16)in[(((b << 9) + s) << 9) + (h << 6) + j];
}
// [B,H,S,dk] f32 -> [B,S,512] f16
__global__ void merge_heads_f16_kernel(const float* __restrict__ in,
                                       _Float16* __restrict__ out,
                                       long long tot) {
  long long i = (long long)blockIdx.x * blockDim.x + threadIdx.x;
  if (i >= tot) return;
  int j = (int)(i & 63);
  int h = (int)((i >> 6) & 7);
  int s = (int)((i >> 9) & 511);
  long long b = i >> 18;
  out[i] = (_Float16)in[(((((b << 3) + h) << 9) + s) << 6) + j];
}

// ------------------------------ orchestration ------------------------------
extern "C" void kernel_launch(void* const* d_in, const int* in_sizes, int n_in,
                              void* d_out, int out_size, void* d_ws,
                              size_t ws_size, hipStream_t stream) {
  (void)in_sizes; (void)n_in; (void)out_size; (void)ws_size;

  const int Bb = 4, S = 512, Dm = 512, Hh = 8, dk = 64, DF = 2048, L = 6,
            V = 32000;
  const int BS = Bb * S;

  const int* src = (const int*)d_in[0];
  const int* tgt = (const int*)d_in[1];
  const float* src_emb = (const float*)d_in[2];
  const float* tgt_emb = (const float*)d_in[3];
  const float* out_w = (const float*)d_in[4];
  const float* out_b = (const float*)d_in[5];
  const float* e_gf = (const float*)d_in[6];
  const float* e_bf = (const float*)d_in[7];
  const float* dgf = (const float*)d_in[8];
  const float* dbf = (const float*)d_in[9];
  const float* e_wq = (const float*)d_in[10];
  const float* e_wk = (const float*)d_in[11];
  const float* e_wv = (const float*)d_in[12];
  const float* e_wo = (const float*)d_in[13];
  const float* d_wq = (const float*)d_in[14];
  const float* d_wk = (const float*)d_in[15];
  const float* d_wv = (const float*)d_in[16];
  const float* d_wo = (const float*)d_in[17];
  const float* d_cwq = (const float*)d_in[18];
  const float* d_cwk = (const float*)d_in[19];
  const float* d_cwv = (const float*)d_in[20];
  const float* d_cwo = (const float*)d_in[21];
  const float* e_w1 = (const float*)d_in[22];
  const float* d_w1 = (const float*)d_in[23];
  const float* e_w2 = (const float*)d_in[24];
  const float* d_w2 = (const float*)d_in[25];
  const float* e_fb1 = (const float*)d_in[26];
  const float* d_fb1 = (const float*)d_in[27];
  const float* e_fb2 = (const float*)d_in[28];
  const float* d_fb2 = (const float*)d_in[29];
  const float* e_g1 = (const float*)d_in[30];
  const float* e_g2 = (const float*)d_in[31];
  const float* d_g1 = (const float*)d_in[32];
  const float* d_g2 = (const float*)d_in[33];
  const float* d_g3 = (const float*)d_in[34];
  const float* e_n1 = (const float*)d_in[35];
  const float* e_n2 = (const float*)d_in[36];
  const float* d_n1 = (const float*)d_in[37];
  const float* d_n2 = (const float*)d_in[38];
  const float* d_n3 = (const float*)d_in[39];

  // ------------------------- scratch carve (~160 MB) -----------------------
  char* base = (char*)d_ws;
  size_t off = 0;
  auto takeF = [&](size_t n) { float* r = (float*)(base + off); off += n * 4; return r; };
  auto takeH = [&](size_t n) { _Float16* r = (_Float16*)(base + off); off += n * 2; return r; };
  const size_t BSD = (size_t)BS * Dm;          // 1,048,576
  const size_t SCO = (size_t)Bb * Hh * S * S;  // 8,388,608
  float* xe = takeF(BSD);
  float* yd = takeF(BSD);
  float* mem = takeF(BSD);
  float* t0 = takeF(BSD);
  float* t1 = takeF(BSD);
  float* t2 = takeF(BSD);
  float* t3 = takeF(BSD);
  float* ho = takeF(BSD);
  float* sco = takeF(SCO);
  float* mid = takeF((size_t)BS * DF);
  _Float16* xe16 = takeH(BSD);
  _Float16* yd16 = takeH(BSD);
  _Float16* mem16 = takeH(BSD);
  _Float16* t316 = takeH(BSD);
  _Float16* t016 = takeH(BSD);
  _Float16* hq16 = takeH(BSD);
  _Float16* hk16 = takeH(BSD);
  _Float16* hv16 = takeH(BSD);
  _Float16* sc16 = takeH(SCO);
  _Float16* mid16 = takeH((size_t)BS * DF);
  _Float16* wB16 = takeH((size_t)Dm * V);  // weight BT staging (max 512x32000)

  auto egrid = [](long long n) { return dim3((unsigned)((n + 255) / 256)); };

  auto cvtT = [&](const float* w, int kshift, int N) {
    long long total = (long long)N << kshift;
    cvtT_kernel<<<egrid(total), 256, 0, stream>>>(w, wB16, kshift, N, total);
  };
  auto gemm = [&](const _Float16* A, const _Float16* Bt, const float* bias,
                  float* C, _Float16* C16, int M, int N, int K, int batch,
                  long long sA, long long sB, long long sC, int relu) {
    dim3 grid(N / BN, M / BM, batch);
    wmma_gemm_kernel<<<grid, 256, 0, stream>>>(A, Bt, bias, C, C16, M, N, K,
                                               sA, sB, sC, relu);
  };
  auto addln = [&](const float* x, const float* a, const float* g,
                   const float* bt, float* out, _Float16* out16) {
    add_ln_kernel<<<dim3(BS), 256, 0, stream>>>(x, a, g, bt, out, out16);
  };

  const long long tot = (long long)BS * Dm;
  auto attention = [&](const _Float16* qin16, const _Float16* kvin16,
                       const float* wq, const float* wk, const float* wv,
                       const float* wo, const int* tok, int causal,
                       float* outb) {
    cvtT(wq, 9, Dm);
    gemm(qin16, wB16, nullptr, t0, nullptr, BS, Dm, Dm, 1, 0, 0, 0, 0);
    cvtT(wk, 9, Dm);
    gemm(kvin16, wB16, nullptr, t1, nullptr, BS, Dm, Dm, 1, 0, 0, 0, 0);
    cvtT(wv, 9, Dm);
    gemm(kvin16, wB16, nullptr, t2, nullptr, BS, Dm, Dm, 1, 0, 0, 0, 0);
    split_heads_f16_kernel<<<egrid(tot), 256, 0, stream>>>(t0, hq16, tot);
    split_heads_f16_kernel<<<egrid(tot), 256, 0, stream>>>(t1, hk16, tot);
    split_headsT_f16_kernel<<<egrid(tot), 256, 0, stream>>>(t2, hv16, tot);
    // scores[b,h] = Qh (Sxdk) * Kh^T  (BT = Kh as [S,dk])
    gemm(hq16, hk16, nullptr, sco, nullptr, S, S, dk, Bb * Hh,
         (long long)S * dk, (long long)S * dk, (long long)S * S, 0);
    softmax_mask_kernel<<<dim3(Bb * Hh * S), 256, 0, stream>>>(
        sco, sc16, tok, causal, 0.125f /* 1/sqrt(dk) */);
    // out[b,h] = P (SxS) * Vh  (BT = VhT as [dk,S])
    gemm(sc16, hv16, nullptr, ho, nullptr, S, dk, S, Bb * Hh,
         (long long)S * S, (long long)dk * S, (long long)S * dk, 0);
    merge_heads_f16_kernel<<<egrid(tot), 256, 0, stream>>>(ho, t016, tot);
    cvtT(wo, 9, Dm);
    gemm(t016, wB16, nullptr, outb, nullptr, BS, Dm, Dm, 1, 0, 0, 0, 0);
  };

  // ------------------------------ encoder ----------------------------------
  embed_kernel<<<egrid(tot), 256, 0, stream>>>(src_emb, src, xe, xe16, tot);
  for (int i = 0; i < L; ++i) {
    const size_t wO = (size_t)i * Dm * Dm;
    attention(xe16, xe16, e_wq + wO, e_wk + wO, e_wv + wO, e_wo + wO, src, 0, t3);
    addln(xe, t3, e_g1 + (size_t)i * Dm, e_n1 + (size_t)i * Dm, xe, xe16);
    cvtT(e_w1 + (size_t)i * Dm * DF, 9, DF);
    gemm(xe16, wB16, e_fb1 + (size_t)i * DF, mid, mid16, BS, DF, Dm, 1, 0, 0, 0, 1);
    cvtT(e_w2 + (size_t)i * DF * Dm, 11, Dm);
    gemm(mid16, wB16, e_fb2 + (size_t)i * Dm, t3, nullptr, BS, Dm, DF, 1, 0, 0, 0, 0);
    addln(xe, t3, e_g2 + (size_t)i * Dm, e_n2 + (size_t)i * Dm, xe, xe16);
  }
  addln(xe, nullptr, e_gf, e_bf, mem, mem16);

  // ------------------------------ decoder ----------------------------------
  embed_kernel<<<egrid(tot), 256, 0, stream>>>(tgt_emb, tgt, yd, yd16, tot);
  for (int i = 0; i < L; ++i) {
    const size_t wO = (size_t)i * Dm * Dm;
    attention(yd16, yd16, d_wq + wO, d_wk + wO, d_wv + wO, d_wo + wO, tgt, 1, t3);
    addln(yd, t3, d_g1 + (size_t)i * Dm, d_n1 + (size_t)i * Dm, yd, yd16);
    attention(yd16, mem16, d_cwq + wO, d_cwk + wO, d_cwv + wO, d_cwo + wO, src, 0, t3);
    addln(yd, t3, d_g2 + (size_t)i * Dm, d_n2 + (size_t)i * Dm, yd, yd16);
    cvtT(d_w1 + (size_t)i * Dm * DF, 9, DF);
    gemm(yd16, wB16, d_fb1 + (size_t)i * DF, mid, mid16, BS, DF, Dm, 1, 0, 0, 0, 1);
    cvtT(d_w2 + (size_t)i * DF * Dm, 11, Dm);
    gemm(mid16, wB16, d_fb2 + (size_t)i * Dm, t3, nullptr, BS, Dm, DF, 1, 0, 0, 0, 0);
    addln(yd, t3, d_g3 + (size_t)i * Dm, d_n3 + (size_t)i * Dm, yd, yd16);
  }
  addln(yd, nullptr, dgf, dbf, t3, t316);

  // ------------------------- vocab projection ------------------------------
  cvtT(out_w, 9, V);
  gemm(t316, wB16, out_b, (float*)d_out, nullptr, BS, V, Dm, 1, 0, 0, 0, 0);
}
